// PointTransformerV2_2508260901463
// MI455X (gfx1250) — matrix operand order repaired and checked by reading, not compiled, not run on hardware
//
#include <hip/hip_runtime.h>
#include <hip/hip_bf16.h>
#include <stdint.h>

// ---------------------------------------------------------------------------
// PointTransformerV2 grouped-vector-attention forward, CDNA5 (gfx1250) wave32.
// N=50000 points, S=16 neighbours, C=96 channels, G=6 groups.
// All GEMMs run on v_wmma_f32_16x16x32_f16 (M=16 rows = one point's 16
// neighbours, or one 16-row tile of points for QKV).
// ---------------------------------------------------------------------------

typedef __attribute__((ext_vector_type(16))) _Float16 v16h;
typedef __attribute__((ext_vector_type(8)))  float    v8f;

#define NPTS 50000
#define SNB  16
#define CCH  96
#define GGRP 6
#define RSQ  0.99999499988750f   // 1/sqrt(1+1e-5)

union FragAB {                 // one WMMA A or B operand (16 halves / lane)
    v16h     h;
    _Float16 e[16];
    uint4    q[2];
    uint32_t u[8];
};

union Row8 { float4 v[2]; float f[8]; };

__device__ __forceinline__ v8f wmma_f16(v16h a, v16h b, v8f c) {
    return __builtin_amdgcn_wmma_f32_16x16x32_f16(false, a, false, b,
                                                  (short)0, c, false, false);
}

// Packed B fragment: frag[tile][lane][8 dwords]; per-lane 32B contiguous.
__device__ __forceinline__ v16h load_bfrag(const uint32_t* __restrict__ frag,
                                           int tile, int lane) {
    FragAB f;
    const uint4* p = (const uint4*)(frag + (size_t)(tile * 32 + lane) * 8);
    f.q[0] = p[0];
    f.q[1] = p[1];
    return f.h;
}

// ---------------------------------------------------------------------------
// Kernel 0: pack weight matrices into WMMA B-fragment layout (f32 -> f16).
// Tiles: 0..53 = Wq/Wk/Wv (18 each: kk*6+j), 54..71 = Wp2, 72..74 = Ww1 (padded
// to 16 output cols).  B layout (16-bit, 32x16 KxN): lane<16 -> col=lane,
// K=0..7,16..23; lane>=16 -> col=lane-16, K=8..15,24..31 (table order).
// ---------------------------------------------------------------------------
__global__ void prep_frags_kernel(const float* __restrict__ Wq,
                                  const float* __restrict__ Wk,
                                  const float* __restrict__ Wv,
                                  const float* __restrict__ Wp2,
                                  const float* __restrict__ Ww1,
                                  uint32_t* __restrict__ frag) {
    int tile = blockIdx.x;
    int lane = threadIdx.x & 31;
    const float* W;
    int outDim = CCH, kk, j;
    if (tile < 54) {
        int m = tile / 18, rem = tile % 18;
        W = (m == 0) ? Wq : (m == 1) ? Wk : Wv;
        kk = rem / 6; j = rem % 6;
    } else if (tile < 72) {
        int rem = tile - 54;
        W = Wp2; kk = rem / 6; j = rem % 6;
    } else {
        W = Ww1; outDim = GGRP; kk = tile - 72; j = 0;
    }
    int sl = lane & 15, half = lane >> 4, koff = half * 8;
    int col = j * 16 + sl;
    bool valid = (col < outDim);
    uint32_t out8[8];
#pragma unroll
    for (int v = 0; v < 8; ++v) {
        int r = kk * 32 + koff + 2 * v + (v >= 4 ? 8 : 0);
        float lo = valid ? W[(size_t)r * outDim + col] : 0.0f;
        float hi = valid ? W[(size_t)(r + 1) * outDim + col] : 0.0f;
        union { uint32_t u; _Float16 hh[2]; } pk;
        pk.hh[0] = (_Float16)lo; pk.hh[1] = (_Float16)hi;
        out8[v] = pk.u;
    }
    uint4* dst = (uint4*)(frag + (size_t)(tile * 32 + lane) * 8);
    dst[0] = make_uint4(out8[0], out8[1], out8[2], out8[3]);
    dst[1] = make_uint4(out8[4], out8[5], out8[6], out8[7]);
}

// ---------------------------------------------------------------------------
// Kernel 1: q/k/v = epilogue(feat @ W + b).  One wave per 16-row tile,
// 3125 tiles.  q,k get bn(scale,shift)+relu, v is plain affine.
// A layout (16-bit 16x32): lane half selects K-offset 0/8; halves 0..7 hold
// K=koffA..koffA+7, halves 8..15 hold K=16+koffA..16+koffA+7.
// ---------------------------------------------------------------------------
__global__ __launch_bounds__(256) void qkv_kernel(
        const float* __restrict__ feat,
        const float* __restrict__ bq, const float* __restrict__ gq, const float* __restrict__ betaq,
        const float* __restrict__ bk, const float* __restrict__ gk, const float* __restrict__ betak,
        const float* __restrict__ bv,
        const uint32_t* __restrict__ frag,
        float* __restrict__ qout, float* __restrict__ kout, float* __restrict__ vout) {
    int wave = threadIdx.x >> 5;
    int lane = threadIdx.x & 31;
    int t = blockIdx.x * 8 + wave;
    if (t >= NPTS / 16) return;
    int sl = lane & 15, half = lane >> 4, koffA = half * 8;
    const float* frow = feat + (size_t)(t * 16 + sl) * CCH;

    FragAB aF[3];
#pragma unroll
    for (int kk = 0; kk < 3; ++kk) {
        Row8 r0, r1;
        r0.v[0] = *(const float4*)(frow + kk * 32 + koffA);
        r0.v[1] = *(const float4*)(frow + kk * 32 + koffA + 4);
        r1.v[0] = *(const float4*)(frow + kk * 32 + 16 + koffA);
        r1.v[1] = *(const float4*)(frow + kk * 32 + 16 + koffA + 4);
#pragma unroll
        for (int i = 0; i < 8; ++i) {
            aF[kk].e[i]     = (_Float16)r0.f[i];
            aF[kk].e[8 + i] = (_Float16)r1.f[i];
        }
    }

#pragma unroll
    for (int m = 0; m < 3; ++m) {
        const float* bias  = (m == 0) ? bq : (m == 1) ? bk : bv;
        const float* gamma = (m == 0) ? gq : (m == 1) ? gk : nullptr;
        const float* beta  = (m == 0) ? betaq : (m == 1) ? betak : nullptr;
        float* outp        = (m == 0) ? qout : (m == 1) ? kout : vout;
#pragma unroll
        for (int j = 0; j < 6; ++j) {
            v8f acc = {};
#pragma unroll
            for (int kk = 0; kk < 3; ++kk)
                acc = wmma_f16(aF[kk].h, load_bfrag(frag, m * 18 + kk * 6 + j, lane), acc);
            int c = j * 16 + sl;
            float b = bias[c];
            float sc = 1.0f, be = 0.0f;
            if (gamma) { sc = gamma[c] * RSQ; be = beta[c]; }
#pragma unroll
            for (int v = 0; v < 8; ++v) {
                float y = acc[v] + b;
                if (gamma) { y = y * sc + be; y = fmaxf(y, 0.0f); }
                outp[(size_t)(t * 16 + v + half * 8) * CCH + c] = y;  // C layout: M=v(+8), N=c
            }
        }
    }
}

// ---------------------------------------------------------------------------
// Kernel 2: fused attention.  One wave per point (8 waves / block, 6250 blocks,
// exact).  Per point: peb GEMM (16x96x96, 18 WMMAs), relation built into
// LDS (C-layout -> A-layout transpose), weight-encoding GEMM (3 WMMAs),
// Ww2 + softmax over s via shfl, grouped weighted sum (tile j == group g).
// ---------------------------------------------------------------------------
struct __align__(16) WaveScratch {
    _Float16 rel[16 * CCH];   // relation, row-major [s][c], f16 (3072 B)
    float    r1[16 * GGRP];   // relu(bn(relation@Ww1))       (384 B)
    float    w[16 * GGRP];    // softmax weights              (384 B)
};

__global__ __launch_bounds__(256) void attn_kernel(
        const float* __restrict__ coord, const int* __restrict__ refidx,
        const float* __restrict__ qb, const float* __restrict__ kb, const float* __restrict__ vb,
        const uint32_t* __restrict__ frag,
        const float* __restrict__ Wp1, const float* __restrict__ bp1,
        const float* __restrict__ gp, const float* __restrict__ betap,
        const float* __restrict__ bp2,
        const float* __restrict__ bw1, const float* __restrict__ gw, const float* __restrict__ betaw,
        const float* __restrict__ Ww2, const float* __restrict__ bw2,
        float* __restrict__ out) {
    __shared__ WaveScratch smem[8];
    int wave = threadIdx.x >> 5;
    int lane = threadIdx.x & 31;
    int n = blockIdx.x * 8 + wave;                 // 6250*8 == 50000, no tail
    int sl = lane & 15, half = lane >> 4, koffA = half * 8;
    WaveScratch& ws = smem[wave];

    // --- neighbour index + relative position for row s = sl ---
    int myIdx = refidx[(size_t)n * SNB + sl];
    float px = coord[(size_t)myIdx * 3 + 0] - coord[(size_t)n * 3 + 0];
    float py = coord[(size_t)myIdx * 3 + 1] - coord[(size_t)n * 3 + 1];
    float pz = coord[(size_t)myIdx * 3 + 2] - coord[(size_t)n * 3 + 2];

    // --- h1 = relu(bn(pos @ Wp1 + bp1)) packed directly into A-fragments ---
    FragAB aF[3];
#pragma unroll
    for (int kk = 0; kk < 3; ++kk) {
#pragma unroll
        for (int run = 0; run < 2; ++run) {
            int kbv = kk * 32 + run * 16 + koffA;
            Row8 w0, w1, w2, bb, gg, tt;
            w0.v[0] = *(const float4*)(Wp1 + 0 * CCH + kbv);
            w0.v[1] = *(const float4*)(Wp1 + 0 * CCH + kbv + 4);
            w1.v[0] = *(const float4*)(Wp1 + 1 * CCH + kbv);
            w1.v[1] = *(const float4*)(Wp1 + 1 * CCH + kbv + 4);
            w2.v[0] = *(const float4*)(Wp1 + 2 * CCH + kbv);
            w2.v[1] = *(const float4*)(Wp1 + 2 * CCH + kbv + 4);
            bb.v[0] = *(const float4*)(bp1 + kbv);   bb.v[1] = *(const float4*)(bp1 + kbv + 4);
            gg.v[0] = *(const float4*)(gp + kbv);    gg.v[1] = *(const float4*)(gp + kbv + 4);
            tt.v[0] = *(const float4*)(betap + kbv); tt.v[1] = *(const float4*)(betap + kbv + 4);
#pragma unroll
            for (int i = 0; i < 8; ++i) {
                float val = px * w0.f[i] + py * w1.f[i] + pz * w2.f[i] + bb.f[i];
                val = val * (gg.f[i] * RSQ) + tt.f[i];
                val = fmaxf(val, 0.0f);
                aF[kk].e[run * 8 + i] = (_Float16)val;
            }
        }
    }

    // --- peb = h1 @ Wp2 + bp2 : 18 WMMAs, acc[j] holds C-layout 16x16 tile ---
    v8f acc[6];
#pragma unroll
    for (int j = 0; j < 6; ++j) {
        v8f a = {};
#pragma unroll
        for (int kk = 0; kk < 3; ++kk)
            a = wmma_f16(aF[kk].h, load_bfrag(frag, 54 + kk * 6 + j, lane), a);
        float bp = bp2[j * 16 + sl];
#pragma unroll
        for (int v = 0; v < 8; ++v) a[v] += bp;
        acc[j] = a;
    }

    // --- broadcast the 8 neighbour indices owned by this lane's C-rows ---
    int sIdx[8];
#pragma unroll
    for (int v = 0; v < 8; ++v) sIdx[v] = __shfl(myIdx, v + half * 8, 32);

    // --- relation = kg - q + peb  ->  LDS (f16, row-major);  acc += vg ---
#pragma unroll
    for (int j = 0; j < 6; ++j) {
        int c = j * 16 + sl;
        float qc = qb[(size_t)n * CCH + c];
#pragma unroll
        for (int v = 0; v < 8; ++v) {
            float kg  = kb[(size_t)sIdx[v] * CCH + c];
            float rel = kg - qc + acc[j][v];
            ws.rel[(v + half * 8) * CCH + c] = (_Float16)rel;
            acc[j][v] += vb[(size_t)sIdx[v] * CCH + c];   // vg + peb
        }
    }
    __syncthreads();

    // --- LDS -> A-layout fragments of relation, GEMM with padded Ww1 ---
    FragAB rA[3];
#pragma unroll
    for (int kk = 0; kk < 3; ++kk) {
        const _Float16* rp = ws.rel + sl * CCH + kk * 32 + koffA;
        rA[kk].q[0] = *(const uint4*)(rp);
        rA[kk].q[1] = *(const uint4*)(rp + 16);
    }
    v8f wacc = {};
#pragma unroll
    for (int kk = 0; kk < 3; ++kk)
        wacc = wmma_f16(rA[kk].h, load_bfrag(frag, 72 + kk, lane), wacc);

    int g = sl;                          // column of the weight tile
    if (g < GGRP) {
        float sc = gw[g] * RSQ, b1 = bw1[g], bt = betaw[g];
#pragma unroll
        for (int v = 0; v < 8; ++v) {
            float y = (wacc[v] + b1) * sc + bt;
            ws.r1[(v + half * 8) * GGRP + g] = fmaxf(y, 0.0f);
        }
    }
    __syncthreads();

    // --- logits = r1 @ Ww2 + bw2 ; softmax over s (16-lane shfl groups) ---
    float r[GGRP];
#pragma unroll
    for (int gg2 = 0; gg2 < GGRP; ++gg2) r[gg2] = ws.r1[sl * GGRP + gg2];
    int ip = myIdx + 1;
    float msk = (ip > 0) ? 1.0f : ((ip < 0) ? -1.0f : 0.0f);   // sign(idx+1)
#pragma unroll
    for (int go = 0; go < GGRP; ++go) {
        float lg = bw2[go];
#pragma unroll
        for (int gi = 0; gi < GGRP; ++gi) lg += r[gi] * Ww2[gi * GGRP + go];
        float mx = lg;
        mx = fmaxf(mx, __shfl_xor(mx, 8, 16));
        mx = fmaxf(mx, __shfl_xor(mx, 4, 16));
        mx = fmaxf(mx, __shfl_xor(mx, 2, 16));
        mx = fmaxf(mx, __shfl_xor(mx, 1, 16));
        float e = __expf(lg - mx);
        float sm = e;
        sm += __shfl_xor(sm, 8, 16);
        sm += __shfl_xor(sm, 4, 16);
        sm += __shfl_xor(sm, 2, 16);
        sm += __shfl_xor(sm, 1, 16);
        float wv = (e / sm) * msk;
        if (half == 0) ws.w[sl * GGRP + go] = wv;
    }
    __syncthreads();

    // --- out[n, g*16+i] = sum_s (vg+peb)[s, g*16+i] * w[s,g] ; tile j == g ---
#pragma unroll
    for (int j = 0; j < 6; ++j) {
        float p = 0.0f;
#pragma unroll
        for (int v = 0; v < 8; ++v)
            p += acc[j][v] * ws.w[(v + half * 8) * GGRP + j];
        p += __shfl_xor(p, 16, 32);                    // combine s halves
        if (half == 0) out[(size_t)n * CCH + j * 16 + sl] = p;
    }
}

// ---------------------------------------------------------------------------
// Host side
// ---------------------------------------------------------------------------
extern "C" void kernel_launch(void* const* d_in, const int* in_sizes, int n_in,
                              void* d_out, int out_size, void* d_ws, size_t ws_size,
                              hipStream_t stream) {
    const float* feat   = (const float*)d_in[0];
    const float* coord  = (const float*)d_in[1];
    const int*   ridx   = (const int*)d_in[2];
    const float* Wq = (const float*)d_in[3],  *bq = (const float*)d_in[4];
    const float* gq = (const float*)d_in[5],  *betaq = (const float*)d_in[6];
    const float* Wk = (const float*)d_in[7],  *bk = (const float*)d_in[8];
    const float* gk = (const float*)d_in[9],  *betak = (const float*)d_in[10];
    const float* Wv = (const float*)d_in[11], *bv = (const float*)d_in[12];
    const float* Wp1 = (const float*)d_in[13], *bp1 = (const float*)d_in[14];
    const float* gp = (const float*)d_in[15],  *betap = (const float*)d_in[16];
    const float* Wp2 = (const float*)d_in[17], *bp2 = (const float*)d_in[18];
    const float* Ww1 = (const float*)d_in[19], *bw1 = (const float*)d_in[20];
    const float* gw = (const float*)d_in[21],  *betaw = (const float*)d_in[22];
    const float* Ww2 = (const float*)d_in[23], *bw2 = (const float*)d_in[24];

    const size_t QKV_BYTES = (size_t)NPTS * CCH * sizeof(float);  // 19.2 MB
    uint8_t* ws = (uint8_t*)d_ws;
    float*    qbuf = (float*)(ws + 0 * QKV_BYTES);
    float*    kbuf = (float*)(ws + 1 * QKV_BYTES);
    float*    vbuf = (float*)(ws + 2 * QKV_BYTES);
    uint32_t* frag = (uint32_t*)(ws + 3 * QKV_BYTES);             // 75 KiB of B-frags

    prep_frags_kernel<<<75, 32, 0, stream>>>(Wq, Wk, Wv, Wp2, Ww1, frag);
    qkv_kernel<<<(NPTS / 16 + 7) / 8, 256, 0, stream>>>(
        feat, bq, gq, betaq, bk, gk, betak, bv, frag, qbuf, kbuf, vbuf);
    attn_kernel<<<NPTS / 8, 256, 0, stream>>>(
        coord, ridx, qbuf, kbuf, vbuf, frag,
        Wp1, bp1, gp, betap, bp2, bw1, gw, betaw, Ww2, bw2, (float*)d_out);
}
